// MultiHeadAttention_13597866459930
// MI455X (gfx1250) — matrix operand order repaired and checked by reading, not compiled, not run on hardware
//
#include <hip/hip_runtime.h>

// ---------------------------------------------------------------------------
// CDNA5 (gfx1250, wave32) multi-head attention.
// bf16 WMMA (v_wmma_f32_16x16x32_bf16) + Tensor Data Mover tile staging with
// TENSORcnt-based double buffering. One cheap f32->bf16 convert prepass makes
// every GEMM tile a pure TDM descriptor load (incl. LDS row padding via the
// descriptor's pad_enable/pad_interval/pad_amount fields).
// ---------------------------------------------------------------------------

typedef __attribute__((ext_vector_type(16))) __bf16       v16bf;
typedef __attribute__((ext_vector_type(8)))  float        v8f;
typedef __attribute__((ext_vector_type(4)))  unsigned int v4u;
typedef __attribute__((ext_vector_type(8)))  int          v8i;
typedef __attribute__((ext_vector_type(4)))  int          v4i;

#define D_MODELc 512
#define SEQ_Nc   2048
#define BATCHc   4
#define HEADSc   8
#define DKc      64
#define M_TOTALc (BATCHc * SEQ_Nc)   // 8192

#define TMc 128
#define TNc 128
#define TKc 32
#define APADc 8                      // LDS row pad (bf16 elems): 16B per row

#if __has_builtin(__builtin_amdgcn_tensor_load_to_lds) && \
    __has_builtin(__builtin_amdgcn_s_wait_tensorcnt)
#define HAVE_TDM 1
#else
#define HAVE_TDM 0
#endif

struct B16Pair { uint4 lo, hi; };    // 32 bytes == one v16bf fragment

static __device__ __forceinline__ __bf16 to_bf16(float f) {
  unsigned u = __builtin_bit_cast(unsigned, f);
  u += 0x7FFFu + ((u >> 16) & 1u);                  // round-to-nearest-even
  unsigned short h = (unsigned short)(u >> 16);
  return __builtin_bit_cast(__bf16, h);
}

static __device__ __forceinline__ v16bf load16(const __bf16* p) {
  B16Pair t;
  t.lo = *(const uint4*)p;
  t.hi = *(const uint4*)(p + 8);
  return __builtin_bit_cast(v16bf, t);
}
static __device__ __forceinline__ v16bf load8x2(const __bf16* p0, const __bf16* p1) {
  B16Pair t;
  t.lo = *(const uint4*)p0;
  t.hi = *(const uint4*)p1;
  return __builtin_bit_cast(v16bf, t);
}

static __device__ __forceinline__ v8f wmma_bf16(v16bf a, v16bf b, v8f c) {
  return __builtin_amdgcn_wmma_f32_16x16x32_bf16(
      false, a, false, b, (short)0, c, false, false);
}

#if HAVE_TDM
// Workgroup-relative LDS byte address for a TDM descriptor.
#define LDS_ADDR_U32(p) \
  ((unsigned)(__SIZE_TYPE__)(__attribute__((address_space(3))) const void*)(p))

// 2D bf16 tile load: tensor row-major [dim1][stride_e], tile (tile_h rows x
// tile_w elems) starting at gsrc, destination LDS rows padded by pad codes.
// pad_i: pad after 2^pad_i * 8 bytes; pad_a: (pad_a+1) DWORDs of pad.
static __device__ __forceinline__ void tdm_load_2d_bf16(
    const __bf16* gsrc, unsigned lds_byte_off,
    unsigned tile_w, unsigned tile_h,
    unsigned stride_e, unsigned dim0, unsigned dim1,
    unsigned pad_i, unsigned pad_a)
{
  unsigned long long ga = (unsigned long long)(__SIZE_TYPE__)gsrc;
  v4u g0 = {0u, 0u, 0u, 0u};
  g0[0] = 1u;                                            // count=1 (user D#)
  g0[1] = lds_byte_off;                                  // lds_addr
  g0[2] = (unsigned)ga;                                  // global_addr[31:0]
  g0[3] = (unsigned)((ga >> 32) & 0x01FFFFFFull) | (2u << 30);  // addr hi | type=2
  v8i g1 = {0, 0, 0, 0, 0, 0, 0, 0};
  g1[0] = (int)((1u << 16) | (1u << 20) | (pad_i << 22) | (pad_a << 25)); // 2B elems, pad
  g1[1] = (int)((dim0 & 0xFFFFu) << 16);                 // tensor_dim0 lo
  g1[2] = (int)(((dim0 >> 16) & 0xFFFFu) | ((dim1 & 0xFFFFu) << 16));
  g1[3] = (int)(((dim1 >> 16) & 0xFFFFu) | ((tile_w & 0xFFFFu) << 16));
  g1[4] = (int)(tile_h & 0xFFFFu);                       // tile_dim1 (tile_dim2=0)
  g1[5] = (int)stride_e;                                 // tensor_dim0_stride
  v4i z4 = {0, 0, 0, 0};
#if defined(__clang_major__) && __clang_major__ >= 23
  v8i z8 = {0, 0, 0, 0, 0, 0, 0, 0};
  __builtin_amdgcn_tensor_load_to_lds(g0, g1, z4, z4, z8, 0);
#else
  __builtin_amdgcn_tensor_load_to_lds(g0, g1, z4, z4, 0);
#endif
}
#endif  // HAVE_TDM

// ---------------------------------------------------------------------------
// Kernel 0: convert prepass.  x -> bf16 (same layout); W{q,k,v,o} -> bf16
// TRANSPOSED to [n][k] so all GEMM A/B tiles become straight 2D TDM loads.
// grid (4096, 1, 5), block 256.
// ---------------------------------------------------------------------------
__global__ __launch_bounds__(256) void convert_kernel(
    const float* __restrict__ x,
    const float* __restrict__ Wq, const float* __restrict__ Wk,
    const float* __restrict__ Wv, const float* __restrict__ Wo,
    __bf16* __restrict__ xbf,
    __bf16* __restrict__ wtq, __bf16* __restrict__ wtk,
    __bf16* __restrict__ wtv, __bf16* __restrict__ wto)
{
  const int tid = threadIdx.x;
  if (blockIdx.z == 0) {
    size_t i = ((size_t)blockIdx.x * 256 + tid) * 4;
    float4 v = *(const float4*)(x + i);
    xbf[i + 0] = to_bf16(v.x); xbf[i + 1] = to_bf16(v.y);
    xbf[i + 2] = to_bf16(v.z); xbf[i + 3] = to_bf16(v.w);
  } else {
    if (blockIdx.x >= (D_MODELc * D_MODELc) / 256) return;   // 1024 blocks
    const float* W; __bf16* Wt;
    switch (blockIdx.z) {
      case 1:  W = Wq; Wt = wtq; break;
      case 2:  W = Wk; Wt = wtk; break;
      case 3:  W = Wv; Wt = wtv; break;
      default: W = Wo; Wt = wto; break;
    }
    int idx = blockIdx.x * 256 + tid;        // 0..262143
    int k = idx >> 9, n = idx & 511;
    Wt[(size_t)n * D_MODELc + k] = to_bf16(W[idx]);   // coalesced read
  }
}

// ---------------------------------------------------------------------------
// Kernel 1: QKV projection (bf16 A = xbf, bf16 B = Wt[n][k]).
// TDM-staged, TENSORcnt double-buffered. grid (4, 64, 3), block 256 (8 waves).
// ---------------------------------------------------------------------------
__global__ __launch_bounds__(256) void qkv_proj_kernel(
    const __bf16* __restrict__ xbf,
    const __bf16* __restrict__ wtq, const float* __restrict__ bq,
    const __bf16* __restrict__ wtk, const float* __restrict__ bk,
    const __bf16* __restrict__ wtv, const float* __restrict__ bv,
    __bf16* __restrict__ qws, __bf16* __restrict__ kws, __bf16* __restrict__ vws)
{
  __shared__ __bf16 As[2][TMc][TKc + APADc];   // [m][k], TDM-padded rows
  __shared__ __bf16 Bs[2][TNc][TKc + APADc];   // [n][k]

  const int tid  = threadIdx.x;
  const int w    = tid >> 5;
  const int lane = tid & 31;
  const int hf   = (lane >> 4) & 1;
  const int ln   = lane & 15;
  const int wm   = w >> 1;
  const int wn   = w & 1;
  const int m0   = blockIdx.y * TMc;
  const int n0   = blockIdx.x * TNc;

  const __bf16* Wt; const float* bias; __bf16* dst; float scale;
  if (blockIdx.z == 0)      { Wt = wtq; bias = bq; dst = qws; scale = 0.125f; }
  else if (blockIdx.z == 1) { Wt = wtk; bias = bk; dst = kws; scale = 1.0f; }
  else                      { Wt = wtv; bias = bv; dst = vws; scale = 1.0f; }

  const v8f vzero = {0.f,0.f,0.f,0.f,0.f,0.f,0.f,0.f};
  v8f acc[2][4];
#pragma unroll
  for (int i = 0; i < 2; ++i)
#pragma unroll
    for (int j = 0; j < 4; ++j) acc[i][j] = vzero;

#if HAVE_TDM
  if (w == 0) {   // prologue: fill buffer 0
    tdm_load_2d_bf16(xbf + (size_t)m0 * D_MODELc, LDS_ADDR_U32(&As[0][0][0]),
                     TKc, TMc, D_MODELc, D_MODELc, M_TOTALc, 3, 3);
    tdm_load_2d_bf16(Wt + (size_t)n0 * D_MODELc, LDS_ADDR_U32(&Bs[0][0][0]),
                     TKc, TNc, D_MODELc, D_MODELc, D_MODELc, 3, 3);
  }
#endif

  for (int k0 = 0; k0 < D_MODELc; k0 += TKc) {
    const int buf = (k0 >> 5) & 1;
    __syncthreads();                 // prev compute on buf^1 done
#if HAVE_TDM
    if (w == 0) {
      if (k0 + TKc < D_MODELc) {     // pipeline next tile, wait current
        tdm_load_2d_bf16(xbf + (size_t)m0 * D_MODELc + k0 + TKc,
                         LDS_ADDR_U32(&As[buf ^ 1][0][0]),
                         TKc, TMc, D_MODELc, D_MODELc, M_TOTALc, 3, 3);
        tdm_load_2d_bf16(Wt + (size_t)n0 * D_MODELc + k0 + TKc,
                         LDS_ADDR_U32(&Bs[buf ^ 1][0][0]),
                         TKc, TNc, D_MODELc, D_MODELc, D_MODELc, 3, 3);
        __builtin_amdgcn_s_wait_tensorcnt(2);
      } else {
        __builtin_amdgcn_s_wait_tensorcnt(0);
      }
    }
#else
#pragma unroll
    for (int it = 0; it < 2; ++it) {
      int chunk = it * 256 + tid;                  // 512 chunks of 8 bf16
      int r = chunk >> 2, c8 = (chunk & 3) * 8;
      *(uint4*)&As[buf][r][c8] =
          *(const uint4*)(xbf + (size_t)(m0 + r) * D_MODELc + k0 + c8);
      *(uint4*)&Bs[buf][r][c8] =
          *(const uint4*)(Wt + (size_t)(n0 + r) * D_MODELc + k0 + c8);
    }
#endif
    __syncthreads();                 // buf ready

    v16bf af[2];
#pragma unroll
    for (int i = 0; i < 2; ++i) {
      const __bf16* row = &As[buf][wm*32 + i*16 + ln][0];
      af[i] = load8x2(row + hf*8, row + 16 + hf*8);
    }
#pragma unroll
    for (int j = 0; j < 4; ++j) {
      const __bf16* col = &Bs[buf][wn*64 + j*16 + ln][0];
      v16bf bf = load8x2(col + hf*8, col + 16 + hf*8);
#pragma unroll
      for (int i = 0; i < 2; ++i) acc[i][j] = wmma_bf16(af[i], bf, acc[i][j]);
    }
  }

  // Epilogue: bias, scale (1/sqrt(d_k) folded into Q), scatter to [B,H,N,d_k].
#pragma unroll
  for (int i = 0; i < 2; ++i)
#pragma unroll
    for (int j = 0; j < 4; ++j) {
      int ng = n0 + wn*64 + j*16 + ln;
      int h = ng >> 6, d = ng & 63;
      float bval = bias[ng];
#pragma unroll
      for (int r = 0; r < 8; ++r) {
        int m = m0 + wm*32 + i*16 + hf*8 + r;
        int b = m >> 11, t = m & 2047;
        float val = (acc[i][j][r] + bval) * scale;
        dst[((size_t)(b * HEADSc + h) * SEQ_Nc + t) * DKc + d] = to_bf16(val);
      }
    }
}

// ---------------------------------------------------------------------------
// Kernel 2: flash attention per (b,h). K tiles staged by TDM (row-major is
// already the B-fragment layout for Q*K^T); V transposed via VGPRs.
// grid (16, 32), block 256 (8 waves x 16 query rows).
// ---------------------------------------------------------------------------
__global__ __launch_bounds__(256) void attention_kernel(
    const __bf16* __restrict__ qws, const __bf16* __restrict__ kws,
    const __bf16* __restrict__ vws, __bf16* __restrict__ ctx)
{
  __shared__ __bf16 Ks[64][DKc + APADc];         // K tile row-major [key][d]
  __shared__ __bf16 Vs[DKc][DKc + APADc];        // V tile transposed [d][key]
  __shared__ __bf16 Ps[8][16][DKc + APADc];      // per-wave P [row][key]

  const int tid  = threadIdx.x;
  const int w    = tid >> 5;
  const int lane = tid & 31;
  const int hf   = (lane >> 4) & 1;
  const int ln   = lane & 15;
  const int bh   = blockIdx.y;
  const int b    = bh >> 3, h = bh & 7;
  const int t0   = blockIdx.x * 128;

  const __bf16* qp = qws + (size_t)bh * SEQ_Nc * DKc;
  const __bf16* kp = kws + (size_t)bh * SEQ_Nc * DKc;
  const __bf16* vp = vws + (size_t)bh * SEQ_Nc * DKc;

  // Q fragments (16 rows x 64 d_k) resident in VGPRs; Q pre-scaled.
  v16bf qf[2];
  {
    const __bf16* qrow = qp + (size_t)(t0 + w*16 + ln) * DKc;
#pragma unroll
    for (int s = 0; s < 2; ++s)
      qf[s] = load8x2(qrow + s*32 + hf*8, qrow + s*32 + 16 + hf*8);
  }

  const v8f vzero = {0.f,0.f,0.f,0.f,0.f,0.f,0.f,0.f};
  v8f o[4];
#pragma unroll
  for (int dt = 0; dt < 4; ++dt) o[dt] = vzero;
  float mrow[8], lrow[8];
#pragma unroll
  for (int r = 0; r < 8; ++r) { mrow[r] = -1e30f; lrow[r] = 0.f; }

  const int vkey = tid >> 2;          // 0..63
  const int vd0  = (tid & 3) * 16;    // 16 d's per thread

  for (int kt = 0; kt < SEQ_Nc / 64; ++kt) {
    __syncthreads();                  // previous iteration's Ks/Vs/Ps readers done
#if HAVE_TDM
    if (w == 0)
      tdm_load_2d_bf16(kp + (size_t)kt * 64 * DKc, LDS_ADDR_U32(&Ks[0][0]),
                       DKc, 64, DKc, DKc, SEQ_Nc, 4, 3);
#else
    {
      const __bf16* krow = kp + (size_t)(kt*64 + vkey) * DKc + vd0;
      *(uint4*)&Ks[vkey][vd0]     = *(const uint4*)(krow);
      *(uint4*)&Ks[vkey][vd0 + 8] = *(const uint4*)(krow + 8);
    }
#endif
    // Stage V tile transposed into LDS: Vs[d][key].
    {
      const __bf16* vrow = vp + (size_t)(kt*64 + vkey) * DKc + vd0;
      unsigned short tmp[16];
      *(uint4*)&tmp[0] = *(const uint4*)(vrow);
      *(uint4*)&tmp[8] = *(const uint4*)(vrow + 8);
#pragma unroll
      for (int i = 0; i < 16; ++i)
        Vs[vd0 + i][vkey] = __builtin_bit_cast(__bf16, tmp[i]);
    }
    if (kt + 1 < SEQ_Nc / 64)
      __builtin_prefetch(vp + (size_t)((kt+1)*64 + vkey) * DKc + vd0, 0, 1);
#if HAVE_TDM
    if (w == 0) __builtin_amdgcn_s_wait_tensorcnt(0);
#endif
    __syncthreads();                  // Ks (TDM) + Vs visible

    // S = Q * K^T : 16 rows x 64 keys, fragments from LDS.
    v8f s[4];
#pragma unroll
    for (int j = 0; j < 4; ++j) s[j] = vzero;
#pragma unroll
    for (int ks = 0; ks < 2; ++ks) {
#pragma unroll
      for (int j = 0; j < 4; ++j) {
        v16bf bk = load8x2(&Ks[j*16 + ln][ks*32 + hf*8], &Ks[j*16 + ln][ks*32 + 16 + hf*8]);
        s[j] = wmma_bf16(qf[ks], bk, s[j]);
      }
    }

    // Online softmax (row stats replicated across each 16-lane half).
    float mnew[8];
#pragma unroll
    for (int r = 0; r < 8; ++r) mnew[r] = mrow[r];
#pragma unroll
    for (int j = 0; j < 4; ++j)
#pragma unroll
      for (int r = 0; r < 8; ++r) mnew[r] = fmaxf(mnew[r], s[j][r]);
#pragma unroll
    for (int off = 8; off >= 1; off >>= 1)
#pragma unroll
      for (int r = 0; r < 8; ++r)
        mnew[r] = fmaxf(mnew[r], __shfl_xor(mnew[r], off, 32));

    float alpha[8], rs[8];
#pragma unroll
    for (int r = 0; r < 8; ++r) { alpha[r] = __expf(mrow[r] - mnew[r]); mrow[r] = mnew[r]; }
#pragma unroll
    for (int j = 0; j < 4; ++j)
#pragma unroll
      for (int r = 0; r < 8; ++r) s[j][r] = __expf(s[j][r] - mnew[r]);
#pragma unroll
    for (int r = 0; r < 8; ++r) rs[r] = s[0][r] + s[1][r] + s[2][r] + s[3][r];
#pragma unroll
    for (int off = 8; off >= 1; off >>= 1)
#pragma unroll
      for (int r = 0; r < 8; ++r) rs[r] += __shfl_xor(rs[r], off, 32);
#pragma unroll
    for (int r = 0; r < 8; ++r) lrow[r] = lrow[r] * alpha[r] + rs[r];
#pragma unroll
    for (int dt = 0; dt < 4; ++dt)
#pragma unroll
      for (int r = 0; r < 8; ++r) o[dt][r] *= alpha[r];

    // C-layout -> A-layout reshape of P through per-wave LDS.
#pragma unroll
    for (int j = 0; j < 4; ++j)
#pragma unroll
      for (int r = 0; r < 8; ++r)
        Ps[w][hf*8 + r][j*16 + ln] = to_bf16(s[j][r]);
    __syncthreads();                  // Ps visible

    // O += P * V
#pragma unroll
    for (int ks = 0; ks < 2; ++ks) {
      const __bf16* prow = &Ps[w][ln][ks*32];
      v16bf ap = load8x2(prow + hf*8, prow + 16 + hf*8);
#pragma unroll
      for (int dt = 0; dt < 4; ++dt) {
        v16bf bv = load8x2(&Vs[dt*16 + ln][ks*32 + hf*8], &Vs[dt*16 + ln][ks*32 + 16 + hf*8]);
        o[dt] = wmma_bf16(ap, bv, o[dt]);
      }
    }
  }

  // Normalize and write merged-head context [B, N, 512] as bf16.
#pragma unroll
  for (int dt = 0; dt < 4; ++dt) {
    int d = dt*16 + ln;
#pragma unroll
    for (int r = 0; r < 8; ++r) {
      int t = t0 + w*16 + hf*8 + r;
      float val = o[dt][r] / lrow[r];
      ctx[(size_t)(b * SEQ_Nc + t) * D_MODELc + h * DKc + d] = to_bf16(val);
    }
  }
}

// ---------------------------------------------------------------------------
// Kernel 3: output projection (bf16 A = ctx, bf16 B = Wo^T), f32 out + bias.
// TDM-staged, TENSORcnt double-buffered. grid (4, 64), block 256.
// ---------------------------------------------------------------------------
__global__ __launch_bounds__(256) void out_proj_kernel(
    const __bf16* __restrict__ ctx, const __bf16* __restrict__ wto,
    const float* __restrict__ bo, float* __restrict__ out)
{
  __shared__ __bf16 As[2][TMc][TKc + APADc];
  __shared__ __bf16 Bs[2][TNc][TKc + APADc];

  const int tid  = threadIdx.x;
  const int w    = tid >> 5;
  const int lane = tid & 31;
  const int hf   = (lane >> 4) & 1;
  const int ln   = lane & 15;
  const int wm   = w >> 1;
  const int wn   = w & 1;
  const int m0   = blockIdx.y * TMc;
  const int n0   = blockIdx.x * TNc;

  const v8f vzero = {0.f,0.f,0.f,0.f,0.f,0.f,0.f,0.f};
  v8f acc[2][4];
#pragma unroll
  for (int i = 0; i < 2; ++i)
#pragma unroll
    for (int j = 0; j < 4; ++j) acc[i][j] = vzero;

#if HAVE_TDM
  if (w == 0) {
    tdm_load_2d_bf16(ctx + (size_t)m0 * D_MODELc, LDS_ADDR_U32(&As[0][0][0]),
                     TKc, TMc, D_MODELc, D_MODELc, M_TOTALc, 3, 3);
    tdm_load_2d_bf16(wto + (size_t)n0 * D_MODELc, LDS_ADDR_U32(&Bs[0][0][0]),
                     TKc, TNc, D_MODELc, D_MODELc, D_MODELc, 3, 3);
  }
#endif

  for (int k0 = 0; k0 < D_MODELc; k0 += TKc) {
    const int buf = (k0 >> 5) & 1;
    __syncthreads();
#if HAVE_TDM
    if (w == 0) {
      if (k0 + TKc < D_MODELc) {
        tdm_load_2d_bf16(ctx + (size_t)m0 * D_MODELc + k0 + TKc,
                         LDS_ADDR_U32(&As[buf ^ 1][0][0]),
                         TKc, TMc, D_MODELc, D_MODELc, M_TOTALc, 3, 3);
        tdm_load_2d_bf16(wto + (size_t)n0 * D_MODELc + k0 + TKc,
                         LDS_ADDR_U32(&Bs[buf ^ 1][0][0]),
                         TKc, TNc, D_MODELc, D_MODELc, D_MODELc, 3, 3);
        __builtin_amdgcn_s_wait_tensorcnt(2);
      } else {
        __builtin_amdgcn_s_wait_tensorcnt(0);
      }
    }
#else
#pragma unroll
    for (int it = 0; it < 2; ++it) {
      int chunk = it * 256 + tid;
      int r = chunk >> 2, c8 = (chunk & 3) * 8;
      *(uint4*)&As[buf][r][c8] =
          *(const uint4*)(ctx + (size_t)(m0 + r) * D_MODELc + k0 + c8);
      *(uint4*)&Bs[buf][r][c8] =
          *(const uint4*)(wto + (size_t)(n0 + r) * D_MODELc + k0 + c8);
    }
#endif
    __syncthreads();

    v16bf af[2];
#pragma unroll
    for (int i = 0; i < 2; ++i) {
      const __bf16* row = &As[buf][wm*32 + i*16 + ln][0];
      af[i] = load8x2(row + hf*8, row + 16 + hf*8);
    }
#pragma unroll
    for (int j = 0; j < 4; ++j) {
      const __bf16* col = &Bs[buf][wn*64 + j*16 + ln][0];
      v16bf bf = load8x2(col + hf*8, col + 16 + hf*8);
#pragma unroll
      for (int i = 0; i < 2; ++i) acc[i][j] = wmma_bf16(af[i], bf, acc[i][j]);
    }
  }

#pragma unroll
  for (int i = 0; i < 2; ++i)
#pragma unroll
    for (int j = 0; j < 4; ++j) {
      int ng = n0 + wn*64 + j*16 + ln;
      float bval = bo[ng];
#pragma unroll
      for (int r = 0; r < 8; ++r) {
        int m = m0 + wm*32 + i*16 + hf*8 + r;
        out[(size_t)m * D_MODELc + ng] = acc[i][j][r] + bval;
      }
    }
}

// ---------------------------------------------------------------------------
// Host launcher.
// ---------------------------------------------------------------------------
extern "C" void kernel_launch(void* const* d_in, const int* in_sizes, int n_in,
                              void* d_out, int out_size, void* d_ws, size_t ws_size,
                              hipStream_t stream) {
  (void)in_sizes; (void)n_in; (void)out_size; (void)ws_size;

  const float* x  = (const float*)d_in[0];
  const float* Wq = (const float*)d_in[1];
  const float* bq = (const float*)d_in[2];
  const float* Wk = (const float*)d_in[3];
  const float* bk = (const float*)d_in[4];
  const float* Wv = (const float*)d_in[5];
  const float* bv = (const float*)d_in[6];
  const float* Wo = (const float*)d_in[7];
  const float* bo = (const float*)d_in[8];
  float* out = (float*)d_out;

  // Workspace (bf16): xbf | Wt{q,k,v,o} (transposed) | Q | K | V | ctx  ~44 MB
  __bf16* xbf = (__bf16*)d_ws;
  __bf16* wtq = xbf + (size_t)M_TOTALc * D_MODELc;
  __bf16* wtk = wtq + (size_t)D_MODELc * D_MODELc;
  __bf16* wtv = wtk + (size_t)D_MODELc * D_MODELc;
  __bf16* wto = wtv + (size_t)D_MODELc * D_MODELc;
  __bf16* qws = wto + (size_t)D_MODELc * D_MODELc;
  __bf16* kws = qws + (size_t)M_TOTALc * D_MODELc;
  __bf16* vws = kws + (size_t)M_TOTALc * D_MODELc;
  __bf16* cws = vws + (size_t)M_TOTALc * D_MODELc;

  hipLaunchKernelGGL(convert_kernel, dim3(4096, 1, 5), dim3(256), 0, stream,
                     x, Wq, Wk, Wv, Wo, xbf, wtq, wtk, wtv, wto);

  dim3 g1(D_MODELc / TNc, M_TOTALc / TMc, 3);          // (4, 64, 3)
  hipLaunchKernelGGL(qkv_proj_kernel, g1, dim3(256), 0, stream,
                     xbf, wtq, bq, wtk, bk, wtv, bv, qws, kws, vws);

  dim3 g2(SEQ_Nc / 128, BATCHc * HEADSc);              // (16, 32)
  hipLaunchKernelGGL(attention_kernel, g2, dim3(256), 0, stream,
                     qws, kws, vws, cws);

  dim3 g3(D_MODELc / TNc, M_TOTALc / TMc);             // (4, 64)
  hipLaunchKernelGGL(out_proj_kernel, g3, dim3(256), 0, stream,
                     cws, wto, bo, out);
}
